// RNN_Seq_523986010313
// MI455X (gfx1250) — matrix operand order repaired and checked by reading, not compiled, not run on hardware
//
#include <hip/hip_runtime.h>
#include <hip/hip_bf16.h>

#define B_   64
#define T_   1000
#define INP_ 64
#define HID_ 512
#define ACT_ 16
#define NWG_SCAN 16

typedef __attribute__((ext_vector_type(16))) __bf16 v16bf;
typedef __attribute__((ext_vector_type(8)))  __bf16 v8bf;
typedef __attribute__((ext_vector_type(8)))  float  v8f;

__device__ __forceinline__ float sigmoidf_(float x) {
  return 1.0f / (1.0f + __expf(-x));
}

__device__ __forceinline__ v8f wmma_bf16(v16bf a, v16bf b, v8f c) {
  // D = A(16x32 bf16) * B(32x16 bf16) + C(16x16 f32)
  return __builtin_amdgcn_wmma_f32_16x16x32_bf16(false, a, false, b, (short)0, c,
                                                 false, false);
}

// A-fragment gather from an LDS row (row-major, `base` points at start of row M=lane%16).
// 16-bit A layout: lanes 0-15 hold K = {0..7, 16..23}, lanes 16-31 hold K = {8..15, 24..31}.
__device__ __forceinline__ v16bf load_a_frag_lds(const __bf16* base, int ks, int lane) {
  const int hi8 = (lane >> 4) * 8;
  const __bf16* p = base + ks * 32 + hi8;
  v8bf lo = *(const v8bf*)(p);
  v8bf hi = *(const v8bf*)(p + 16);
  v16bf r;
#pragma unroll
  for (int q = 0; q < 8; ++q) { r[q] = lo[q]; r[8 + q] = hi[q]; }
  return r;
}

// ---------------------------------------------------------------------------
// init: reset grid-barrier state, copy y_depression into ping-pong buffer 0
// ---------------------------------------------------------------------------
__global__ void init_kernel(const float* __restrict__ ydep, float* __restrict__ ybuf,
                            unsigned* __restrict__ bars) {
  int i = blockIdx.x * blockDim.x + threadIdx.x;
  if (i < 4) bars[i] = 0u;
  for (int k = i; k < B_ * HID_; k += gridDim.x * blockDim.x) ybuf[k] = ydep[k];
}

// ---------------------------------------------------------------------------
// pack a KxN f32 matrix into the per-lane WMMA B-fragment bf16 layout:
//   out[(((nt*(K/32)+ks)*32 + lane)*16 + j]  with n = nt*16 + lane%16,
//   k = ks*32 + (lane/16)*8 + (j%8) + (j/8)*16.  trans=1 -> source is stored [N][K].
// ---------------------------------------------------------------------------
__global__ void pack_b_kernel(const float* __restrict__ W, __bf16* __restrict__ out,
                              int K, int N, int trans) {
  int idx = blockIdx.x * blockDim.x + threadIdx.x;
  int total = K * N;
  if (idx >= total) return;
  int j      = idx & 15;
  int lane   = (idx >> 4) & 31;
  int blk    = idx >> 9;
  int ksteps = K >> 5;
  int nt = blk / ksteps;
  int ks = blk - nt * ksteps;
  int n = nt * 16 + (lane & 15);
  int k = ks * 32 + (lane >> 4) * 8 + (j & 7) + (j >> 3) * 16;
  float v = trans ? W[(size_t)n * K + k] : W[(size_t)k * N + n];
  out[idx] = (__bf16)v;
}

// ---------------------------------------------------------------------------
// phase1 (fused, fully parallel): for 16 rows of flattened [B*T]:
//   xs = sigmoid(x @ W1^T + b1)  -> LDS (bf16)
//   xtilde = xs @ W_ih           -> global f32
// ---------------------------------------------------------------------------
__global__ void __launch_bounds__(256) phase1_kernel(
    const float* __restrict__ x, const float* __restrict__ b1,
    const __bf16* __restrict__ w1t, const __bf16* __restrict__ wih,
    float* __restrict__ xtilde) {
  __shared__ __bf16 x_lds[16 * INP_];   // 2 KB
  __shared__ __bf16 xs_lds[16 * HID_];  // 16 KB
  const int tid  = threadIdx.x;
  const int lane = tid & 31;
  const int w    = tid >> 5;
  const int r0   = blockIdx.x * 16;

  // stage A: 16 x-rows -> LDS bf16
  {
    int row = tid >> 4;
    int c4  = (tid & 15) * 4;
    float4 v = *(const float4*)&x[(size_t)(r0 + row) * INP_ + c4];
    __bf16* d = &x_lds[row * INP_ + c4];
    d[0] = (__bf16)v.x; d[1] = (__bf16)v.y; d[2] = (__bf16)v.z; d[3] = (__bf16)v.w;
  }
  __syncthreads();

  // stage B: xs = sigmoid(x @ W1^T + b1); each wave covers 4 n-tiles
#pragma unroll
  for (int i = 0; i < 4; ++i) {
    int nt = w * 4 + i;
    v8f acc = {};
#pragma unroll
    for (int ks = 0; ks < 2; ++ks) {  // K = 64
      v16bf va = load_a_frag_lds(&x_lds[(lane & 15) * INP_], ks, lane);
      v16bf vb = *(const v16bf*)&w1t[(size_t)((nt * 2 + ks) * 32 + lane) * 16];
      acc = wmma_bf16(va, vb, acc);
    }
    int n = nt * 16 + (lane & 15);
    float bias = b1[n];
    int mh = (lane >> 4) * 8;
#pragma unroll
    for (int r = 0; r < 8; ++r) {
      float s = sigmoidf_(acc[r] + bias);
      xs_lds[(r + mh) * HID_ + n] = (__bf16)s;
    }
  }
  __syncthreads();

  // stage C: xtilde = xs @ W_ih
#pragma unroll
  for (int i = 0; i < 4; ++i) {
    int nt = w * 4 + i;
    v8f acc = {};
#pragma unroll
    for (int ks = 0; ks < 16; ++ks) {  // K = 512
      v16bf va = load_a_frag_lds(&xs_lds[(lane & 15) * HID_], ks, lane);
      v16bf vb = *(const v16bf*)&wih[(size_t)((nt * 16 + ks) * 32 + lane) * 16];
      acc = wmma_bf16(va, vb, acc);
    }
    int n = nt * 16 + (lane & 15);
    int mh = (lane >> 4) * 8;
#pragma unroll
    for (int r = 0; r < 8; ++r) {
      xtilde[(size_t)(r0 + r + mh) * HID_ + n] = acc[r];
    }
  }
}

// ---------------------------------------------------------------------------
// scan: 16 persistent WGs, WG j owns 32 columns of HID. W_hh slice lives in
// LDS for the whole scan. Per step: pack A = bf16(y*h) into LDS, one
// 64x32x512 bf16 WMMA GEMM per WG, sigmoid + y-update epilogue, grid barrier.
// ---------------------------------------------------------------------------
__global__ void __launch_bounds__(256) scan_kernel(
    const float* __restrict__ hn, const float* __restrict__ xtilde,
    const __bf16* __restrict__ whh, float* __restrict__ ybuf,
    float* __restrict__ newh, float* __restrict__ hnf, float* __restrict__ yf,
    unsigned* __restrict__ bars) {
  __shared__ __bf16 a_lds[B_ * HID_];         // 64 KB: bf16(y*h), full [64x512]
  __shared__ __bf16 w_lds[2 * 16 * 32 * 16];  // 32 KB: this WG's W_hh slice
  const int tid  = threadIdx.x;
  const int lane = tid & 31;
  const int w    = tid >> 5;
  const int j    = blockIdx.x;  // column-slice owner: n in [j*32, j*32+32)
  const int lnt  = w >> 2;      // local n-tile 0..1
  const int mt   = w & 3;       // m-tile 0..3

  // preload resident W_hh slice (packed layout is contiguous per n-tile)
  {
    const uint4* src = (const uint4*)(whh + (size_t)(j * 2) * (16 * 32 * 16));
    uint4* dst = (uint4*)w_lds;
    for (int i = tid; i < (2 * 16 * 32 * 16) / 8; i += 256) dst[i] = src[i];
  }
  unsigned* bar_count = bars;
  volatile unsigned* bar_gen = (volatile unsigned*)(bars + 1);
  __syncthreads();

  for (int t = 0; t < T_; ++t) {
    const float* yb_r = ybuf + (size_t)(t & 1) * (B_ * HID_);
    float*       yb_w = ybuf + (size_t)((t + 1) & 1) * (B_ * HID_);

    // prefetch this step's xtilde slice (one 128B line per batch row)
    if (tid < B_) {
      __builtin_prefetch(&xtilde[((size_t)tid * T_ + t) * HID_ + j * 32], 0, 0);
    }

    // pack A = bf16(y * h_prev), full [64 x 512]
#pragma unroll
    for (int i = 0; i < 32; ++i) {
      int c = tid + i * 256;  // 0..8191 float4 chunks
      int b = c >> 7;
      int n = (c & 127) * 4;
      float4 hv = (t == 0)
          ? *(const float4*)&hn[b * HID_ + n]
          : *(const float4*)&newh[((size_t)b * T_ + (t - 1)) * HID_ + n];
      float4 yv = *(const float4*)&yb_r[b * HID_ + n];
      __bf16* d = &a_lds[b * HID_ + n];
      d[0] = (__bf16)(hv.x * yv.x); d[1] = (__bf16)(hv.y * yv.y);
      d[2] = (__bf16)(hv.z * yv.z); d[3] = (__bf16)(hv.w * yv.w);
    }
    __syncthreads();

    // GEMM over K = 512
    v8f acc = {};
#pragma unroll
    for (int ks = 0; ks < 16; ++ks) {
      v16bf va = load_a_frag_lds(&a_lds[(mt * 16 + (lane & 15)) * HID_], ks, lane);
      v16bf vb = *(const v16bf*)&w_lds[((lnt * 16 + ks) * 32 + lane) * 16];
      acc = wmma_bf16(va, vb, acc);
    }

    // epilogue: h = sigmoid(acc + xtilde), y update (each lane owns its (b,n))
    {
      int n  = j * 32 + lnt * 16 + (lane & 15);
      int mh = (lane >> 4) * 8;
#pragma unroll
      for (int r = 0; r < 8; ++r) {
        int b = mt * 16 + r + mh;
        size_t row = (size_t)b * T_ + t;
        float v = acc[r] + xtilde[row * HID_ + n];
        float h = sigmoidf_(v);
        newh[row * HID_ + n] = h;
        float yo = yb_r[b * HID_ + n];
        float yn = yo + 0.2f * (-(yo - 1.0f) * (1.0f - h) - (yo - 0.25f) * h);
        yb_w[b * HID_ + n] = yn;
      }
    }

    // grid-wide generation barrier across the 16 persistent WGs
    __syncthreads();
    if (tid == 0) {
      __threadfence();
      unsigned prev = atomicAdd(bar_count, 1u);
      if (prev == (unsigned)(NWG_SCAN - 1)) {
        *bar_count = 0u;
        __threadfence();
        atomicAdd((unsigned*)bar_gen, 1u);
      }
      while (*bar_gen < (unsigned)(t + 1)) __builtin_amdgcn_s_sleep(2);
    }
    __syncthreads();
    __threadfence();
  }

  // final carries for this WG's column slice
  for (int i = tid; i < B_ * 32; i += 256) {
    int b = i >> 5;
    int n = j * 32 + (i & 31);
    hnf[b * HID_ + n] = newh[((size_t)b * T_ + (T_ - 1)) * HID_ + n];
    yf[b * HID_ + n]  = ybuf[(size_t)(T_ & 1) * (B_ * HID_) + b * HID_ + n];
  }
}

// ---------------------------------------------------------------------------
// fc2: out = new_h @ W2^T + b2   (M = B*T, N = 16, K = 512)
// ---------------------------------------------------------------------------
__global__ void __launch_bounds__(256) fc2_kernel(
    const float* __restrict__ newh, const __bf16* __restrict__ w2t,
    const float* __restrict__ b2, float* __restrict__ out) {
  __shared__ __bf16 w_lds[16 * 32 * 16];  // 16 KB
  const int tid  = threadIdx.x;
  const int lane = tid & 31;
  const int w    = tid >> 5;
  {
    const uint4* src = (const uint4*)w2t;
    uint4* dst = (uint4*)w_lds;
    for (int i = tid; i < (16 * 32 * 16) / 8; i += 256) dst[i] = src[i];
  }
  __syncthreads();
  const int r0   = blockIdx.x * 128 + w * 16;
  const int arow = r0 + (lane & 15);
  const int hi8  = (lane >> 4) * 8;
  v8f acc = {};
#pragma unroll
  for (int ks = 0; ks < 16; ++ks) {
    const float* p = &newh[(size_t)arow * HID_ + ks * 32 + hi8];
    v16bf va;
#pragma unroll
    for (int q = 0; q < 8; ++q) { va[q] = (__bf16)p[q]; va[8 + q] = (__bf16)p[16 + q]; }
    v16bf vb = *(const v16bf*)&w_lds[(ks * 32 + lane) * 16];
    acc = wmma_bf16(va, vb, acc);
  }
  int a = lane & 15;
  float bias = b2[a];
  int mh = (lane >> 4) * 8;
#pragma unroll
  for (int r = 0; r < 8; ++r) {
    int bt = r0 + r + mh;
    out[(size_t)bt * ACT_ + a] = acc[r] + bias;
  }
}

// ---------------------------------------------------------------------------
extern "C" void kernel_launch(void* const* d_in, const int* in_sizes, int n_in,
                              void* d_out, int out_size, void* d_ws, size_t ws_size,
                              hipStream_t stream) {
  (void)in_sizes; (void)n_in; (void)out_size; (void)ws_size;
  const float* x    = (const float*)d_in[0];
  const float* hn   = (const float*)d_in[1];
  const float* ydep = (const float*)d_in[2];
  const float* W1   = (const float*)d_in[3];
  const float* b1   = (const float*)d_in[4];
  const float* Wih  = (const float*)d_in[5];
  const float* Whh  = (const float*)d_in[6];
  const float* W2   = (const float*)d_in[7];
  const float* b2   = (const float*)d_in[8];
  float* out = (float*)d_out;

  // d_out regions: out [B,T,ACT] | hn_f [B,HID] | new_h [B,T,HID] | y_f [B,HID]
  float* out_main = out;
  float* hnf      = out + (size_t)B_ * T_ * ACT_;
  float* newh     = hnf + (size_t)B_ * HID_;
  float* yf       = newh + (size_t)B_ * T_ * HID_;

  // workspace carve
  char* ws = (char*)d_ws;
  size_t off = 0;
  float*  xtilde = (float*)(ws + off);  off += (size_t)B_ * T_ * HID_ * 4;  // 131 MB
  __bf16* w1t    = (__bf16*)(ws + off); off += (size_t)HID_ * INP_ * 2;
  __bf16* wih    = (__bf16*)(ws + off); off += (size_t)HID_ * HID_ * 2;
  __bf16* whh    = (__bf16*)(ws + off); off += (size_t)HID_ * HID_ * 2;
  __bf16* w2t    = (__bf16*)(ws + off); off += (size_t)HID_ * ACT_ * 2;
  float*  ybuf   = (float*)(ws + off);  off += (size_t)2 * B_ * HID_ * 4;   // ping-pong
  unsigned* bars = (unsigned*)(ws + off); off += 256;

  init_kernel<<<32, 256, 0, stream>>>(ydep, ybuf, bars);
  pack_b_kernel<<<(HID_ * INP_ + 255) / 256, 256, 0, stream>>>(W1, w1t, INP_, HID_, 1);
  pack_b_kernel<<<(HID_ * HID_ + 255) / 256, 256, 0, stream>>>(Wih, wih, HID_, HID_, 0);
  pack_b_kernel<<<(HID_ * HID_ + 255) / 256, 256, 0, stream>>>(Whh, whh, HID_, HID_, 0);
  pack_b_kernel<<<(HID_ * ACT_ + 255) / 256, 256, 0, stream>>>(W2, w2t, HID_, ACT_, 1);
  phase1_kernel<<<(B_ * T_) / 16, 256, 0, stream>>>(x, b1, w1t, wih, xtilde);
  scan_kernel<<<NWG_SCAN, 256, 0, stream>>>(hn, xtilde, whh, ybuf, newh, hnf, yf, bars);
  fc2_kernel<<<(B_ * T_) / 128, 256, 0, stream>>>(newh, w2t, b2, out_main);
}
